// Gate_66803921322557
// MI455X (gfx1250) — matrix operand order repaired and checked by reading, not compiled, not run on hardware
//
#include <hip/hip_runtime.h>

typedef __attribute__((ext_vector_type(16))) __bf16 v16bf;
typedef __attribute__((ext_vector_type(2)))  __bf16 v2bf;
typedef __attribute__((ext_vector_type(8)))  float  v8f;
typedef __attribute__((ext_vector_type(4)))  unsigned int v4u;
typedef __attribute__((ext_vector_type(8)))  int v8i;
typedef __attribute__((ext_vector_type(4)))  int v4i;

#define T_TOK 16384
#define DIM   7168
#define NEXP  256
#define MBLK  128      // tokens per block (8 waves x 16 tokens)
#define KSTEP 32
#define ROWU  40       // ushorts per expert row in LDS (80B: 64B data + 16B pad)

#define HAVE_TDM   __has_builtin(__builtin_amdgcn_tensor_load_to_lds)
#define HAVE_SCHED __has_builtin(__builtin_amdgcn_sched_group_barrier)

// pack two f32 -> one dword of two bf16 (low = a, high = b)
__device__ __forceinline__ unsigned pack_bf16(float a, float b) {
#if __has_builtin(__builtin_amdgcn_cvt_pk_bf16_f32)
  union { v2bf v; unsigned u; } r;
  r.v = __builtin_amdgcn_cvt_pk_bf16_f32(a, b);
  return r.u;
#else
  union { float f; unsigned u; } ca, cb;
  ca.f = a; cb.f = b;
  unsigned ua = ca.u + (0x7FFFu + ((ca.u >> 16) & 1u));
  unsigned ub = cb.u + (0x7FFFu + ((cb.u >> 16) & 1u));
  return (ua >> 16) | (ub & 0xFFFF0000u);
#endif
}

// ---- one-shot W f32 -> bf16 conversion (256*7168 elems, 8 per thread) ----
__global__ void __launch_bounds__(256) wconv(const float* __restrict__ w,
                                             uint4* __restrict__ wbf) {
  const int i = blockIdx.x * blockDim.x + threadIdx.x;   // 229376 threads exactly
  const float4* src = reinterpret_cast<const float4*>(w + (size_t)i * 8);
  float4 a = src[0];
  float4 b = src[1];
  uint4 o;
  o.x = pack_bf16(a.x, a.y);
  o.y = pack_bf16(a.z, a.w);
  o.z = pack_bf16(b.x, b.y);
  o.w = pack_bf16(b.z, b.w);
  wbf[i] = o;
}

#if HAVE_TDM
// TDM: load 256-row x 32-col bf16 tile (row stride DIM elems) into LDS with
// 16-DW data + 4-DW pad per row -> 80B LDS rows (matches ROWU).
__device__ __forceinline__ void tdm_load_tile(unsigned ldsoff, const unsigned short* gsrc) {
  unsigned long long ga = (unsigned long long)(uintptr_t)gsrc;
  v4u g0;
  g0[0] = 1u;                                   // count=1 (valid user descriptor)
  g0[1] = ldsoff;                               // lds_addr (bytes)
  g0[2] = (unsigned)ga;                         // global_addr[31:0]
  g0[3] = (unsigned)((ga >> 32) & 0x01FFFFFFu) | (2u << 30);  // addr[56:32], type=2
  v8i g1;
  g1[0] = (1 << 16) | (1 << 20) | (3 << 22) | (3 << 25);
         // data_size=2B | pad_enable | pad_interval=16DW | pad_amount=4DW
  g1[1] = (KSTEP << 16);                        // tensor_dim0[15:0]=32 at bits[63:48]
  g1[2] = (NEXP << 16);                         // tensor_dim0 hi=0; tensor_dim1[15:0]=256
  g1[3] = (KSTEP << 16);                        // tensor_dim1 hi=0; tile_dim0=32
  g1[4] = NEXP;                                 // tile_dim1=256; tile_dim2=0
  g1[5] = DIM;                                  // tensor_dim0_stride[31:0]=7168
  g1[6] = 0;
  g1[7] = 0;
  v4i z4 = {0, 0, 0, 0};
#if __clang_major__ >= 23
  v8i z8 = {0, 0, 0, 0, 0, 0, 0, 0};
  __builtin_amdgcn_tensor_load_to_lds(g0, g1, z4, z4, z8, 0);
#else
  __builtin_amdgcn_tensor_load_to_lds(g0, g1, z4, z4, 0);
#endif
}
#endif

__global__ void __launch_bounds__(256) gate_gemm(const float* __restrict__ x,
                                                 const unsigned short* __restrict__ wbf,
                                                 float* __restrict__ scores) {
  __shared__ unsigned short ldsW[3][NEXP * ROWU];   // 3 x 20480 bytes (TDM ring)

  const int tid     = threadIdx.x;
  const int wave    = tid >> 5;
  const int lane    = tid & 31;
  const int col     = lane & 15;
  const int halfsel = lane >> 4;
  const int t0      = blockIdx.x * MBLK + wave * 16;

  v8f acc[16];
  const v8f vzero = {0.f, 0.f, 0.f, 0.f, 0.f, 0.f, 0.f, 0.f};
#pragma unroll
  for (int i = 0; i < 16; ++i) acc[i] = vzero;

  // A fragment: lanes 0-15 -> K {0..7,16..23}, lanes 16-31 -> K {8..15,24..31}
  const float* aBase = x + (size_t)(t0 + col) * DIM + halfsel * 8;

  // ---- prefetch first A fragment ----
  float4 areg[4];
  areg[0] = reinterpret_cast<const float4*>(aBase)[0];
  areg[1] = reinterpret_cast<const float4*>(aBase)[1];
  areg[2] = reinterpret_cast<const float4*>(aBase + 16)[0];
  areg[3] = reinterpret_cast<const float4*>(aBase + 16)[1];

#if HAVE_TDM
  if (wave == 0) {
    tdm_load_tile((unsigned)(uintptr_t)&ldsW[0][0], wbf);           // tile 0
    tdm_load_tile((unsigned)(uintptr_t)&ldsW[1][0], wbf + KSTEP);   // tile 1
    __builtin_amdgcn_s_wait_tensorcnt(1);   // tile 0 landed (in-order per wave)
  }
  __syncthreads();
#else
  uint4 wreg[4];   // raw bf16 row copy fallback: thread tid owns expert tid
  {
    const uint4* ws = reinterpret_cast<const uint4*>(wbf + (size_t)tid * DIM);
#pragma unroll
    for (int q = 0; q < 4; ++q) wreg[q] = ws[q];
  }
#endif

  int cur = 0;
  for (int k0 = 0; k0 < DIM; k0 += KSTEP) {
#if !HAVE_TDM
    // ---- store prefetched bf16 W rows into lds[cur] ----
    {
      uint4* dst = reinterpret_cast<uint4*>(&ldsW[cur][tid * ROWU]);
#pragma unroll
      for (int q = 0; q < 4; ++q) dst[q] = wreg[q];
    }
#endif
    // ---- convert prefetched A fragment (registers only) ----
    union { v16bf v; unsigned u[8]; } A;
    A.u[0] = pack_bf16(areg[0].x, areg[0].y);
    A.u[1] = pack_bf16(areg[0].z, areg[0].w);
    A.u[2] = pack_bf16(areg[1].x, areg[1].y);
    A.u[3] = pack_bf16(areg[1].z, areg[1].w);
    A.u[4] = pack_bf16(areg[2].x, areg[2].y);
    A.u[5] = pack_bf16(areg[2].z, areg[2].w);
    A.u[6] = pack_bf16(areg[3].x, areg[3].y);
    A.u[7] = pack_bf16(areg[3].z, areg[3].w);

#if !HAVE_TDM
    __syncthreads();
#endif

    // ---- launch transfers for look-ahead tiles; fly during compute ----
#if HAVE_TDM
    if (k0 + 2 * KSTEP < DIM && wave == 0) {
      int ahead = cur + 2; if (ahead >= 3) ahead -= 3;
      tdm_load_tile((unsigned)(uintptr_t)&ldsW[ahead][0], wbf + k0 + 2 * KSTEP);
    }
#endif
    if (k0 + KSTEP < DIM) {
#if !HAVE_TDM
      const uint4* ws = reinterpret_cast<const uint4*>(wbf + (size_t)tid * DIM + k0 + KSTEP);
#pragma unroll
      for (int q = 0; q < 4; ++q) wreg[q] = ws[q];
#endif
      const float* ap = aBase + k0 + KSTEP;
      areg[0] = reinterpret_cast<const float4*>(ap)[0];
      areg[1] = reinterpret_cast<const float4*>(ap)[1];
      areg[2] = reinterpret_cast<const float4*>(ap + 16)[0];
      areg[3] = reinterpret_cast<const float4*>(ap + 16)[1];
    }

    // ---- 16 expert tiles, depth-4 B-fragment ring from LDS ----
    const unsigned short* bbase = &ldsW[cur][col * ROWU + halfsel * 8];
    union Bfrag { v16bf v; uint4 q[2]; };
    Bfrag B[4];
#pragma unroll
    for (int p = 0; p < 4; ++p) {
      const unsigned short* bp = bbase + p * 16 * ROWU;
      B[p].q[0] = *reinterpret_cast<const uint4*>(bp);
      B[p].q[1] = *reinterpret_cast<const uint4*>(bp + 16);
    }
#pragma unroll
    for (int tile = 0; tile < 16; ++tile) {
      acc[tile] = __builtin_amdgcn_wmma_f32_16x16x32_bf16(
          false, A.v, false, B[tile & 3].v, (short)0, acc[tile], false, false);
      if (tile + 4 < 16) {
        const unsigned short* bp = bbase + (tile + 4) * 16 * ROWU;
        B[tile & 3].q[0] = *reinterpret_cast<const uint4*>(bp);
        B[tile & 3].q[1] = *reinterpret_cast<const uint4*>(bp + 16);
      }
    }

#if HAVE_SCHED
    // Pin the software pipeline: 8 DS reads (tiles 0-3), then
    // (1 WMMA + 2 DS reads for tile t+4) x12, then the last 4 WMMAs.
    // 0x100 = DS read, 0x008 = MFMA/WMMA.
    __builtin_amdgcn_sched_group_barrier(0x100, 8, 0);
#pragma unroll
    for (int i = 0; i < 12; ++i) {
      __builtin_amdgcn_sched_group_barrier(0x008, 1, 0);
      __builtin_amdgcn_sched_group_barrier(0x100, 2, 0);
    }
    __builtin_amdgcn_sched_group_barrier(0x008, 4, 0);
#endif

#if HAVE_TDM
    if (wave == 0) {
      if (k0 + 2 * KSTEP < DIM)      __builtin_amdgcn_s_wait_tensorcnt(1);  // next tile landed
      else if (k0 + KSTEP < DIM)     __builtin_amdgcn_s_wait_tensorcnt(0);  // tail: no look-ahead
    }
    __syncthreads();   // reads of lds[cur] done; next buffer visible to all
    cur = (cur == 2) ? 0 : cur + 1;
#else
    cur ^= 1;
#endif
  }

  // ---- sigmoid + store: C layout: VGPR r -> M = r + 8*halfsel, N = col ----
#pragma unroll
  for (int tile = 0; tile < 16; ++tile) {
    const int e = tile * 16 + col;
#pragma unroll
    for (int r = 0; r < 8; ++r) {
      const int m = r + 8 * halfsel;
      float v = acc[tile][r];
      v = 1.0f / (1.0f + __expf(-v));
      scores[(size_t)(t0 + m) * NEXP + e] = v;
    }
  }
}

__global__ void __launch_bounds__(256) router_topk(const float* __restrict__ scores,
                                                   const float* __restrict__ bias,
                                                   float* __restrict__ wout,
                                                   int* __restrict__ idxout) {
  const int t = blockIdx.x * blockDim.x + threadIdx.x;
  if (t >= T_TOK) return;
  const float* s = scores + (size_t)t * NEXP;
  const float NEG = -3.4e38f;

  float gs[8];
#pragma unroll
  for (int g = 0; g < 8; ++g) {
    float m1 = NEG, m2 = NEG;
    for (int j = 0; j < 32; ++j) {
      const int e = g * 32 + j;
      const float v = s[e] + bias[e];
      if (v > m1) { m2 = m1; m1 = v; }
      else if (v > m2) { m2 = v; }
    }
    gs[g] = m1 + m2;
  }

  bool used[8] = {false, false, false, false, false, false, false, false};
  for (int i = 0; i < 4; ++i) {
    int best = 0; float bv = NEG;
    for (int g = 0; g < 8; ++g)
      if (!used[g] && gs[g] > bv) { bv = gs[g]; best = g; }
    used[best] = true;
  }

  float bval[8], oval[8]; int eidx[8];
#pragma unroll
  for (int i = 0; i < 8; ++i) { bval[i] = NEG; oval[i] = 0.f; eidx[i] = 0; }
  for (int g = 0; g < 8; ++g) {
    if (!used[g]) continue;
    for (int j = 0; j < 32; ++j) {
      const int e = g * 32 + j;
      const float ov = s[e];
      const float v = ov + bias[e];
      if (v > bval[7]) {
        int p = 7;
        while (p > 0 && v > bval[p - 1]) {
          bval[p] = bval[p - 1]; oval[p] = oval[p - 1]; eidx[p] = eidx[p - 1];
          --p;
        }
        bval[p] = v; oval[p] = ov; eidx[p] = e;
      }
    }
  }

  float sum = 0.f;
  for (int i = 0; i < 8; ++i) sum += oval[i];
  const float scale = 2.5f / sum;
  for (int i = 0; i < 8; ++i) {
    wout[(size_t)t * 8 + i]   = oval[i] * scale;
    idxout[(size_t)t * 8 + i] = eidx[i];
  }
}

extern "C" void kernel_launch(void* const* d_in, const int* in_sizes, int n_in,
                              void* d_out, int out_size, void* d_ws, size_t ws_size,
                              hipStream_t stream) {
  const float* x    = (const float*)d_in[0];
  const float* w    = (const float*)d_in[1];
  const float* bias = (const float*)d_in[2];

  float* scores = (float*)d_ws;                                        // 16 MiB
  unsigned short* wbf =
      (unsigned short*)((char*)d_ws + (size_t)T_TOK * NEXP * 4);       // +3.5 MiB bf16 W
  float* wout   = (float*)d_out;
  int*   idxout = (int*)d_out + (size_t)T_TOK * 8;

  wconv<<<(NEXP * DIM / 8) / 256, 256, 0, stream>>>(w, (uint4*)wbf);
  gate_gemm<<<T_TOK / MBLK, 256, 0, stream>>>(x, wbf, scores);
  router_topk<<<(T_TOK + 255) / 256, 256, 0, stream>>>(scores, bias, wout, idxout);
}
